// RT60Prego_18167711662152
// MI455X (gfx1250) — compile-verified, matrix operand order checked by reading, not run on hardware
//
#include <hip/hip_runtime.h>
#include <hip/hip_bf16.h>

// ---------------------------------------------------------------------------
// Problem constants (from the reference)
// ---------------------------------------------------------------------------
#define FS_      16000
#define MWIN     800
#define KFFT     1024
#define HOPSZ    600
#define NBINS    256
#define TFRM     267            // STFT frames per batch
#define BATCH    32
#define TLEN     160000
#define MROWS    (BATCH * TFRM) // 8544 = 534 * 16
#define MTILES   534
#define L_MAXC   13
#define ROWCNT   (NBINS * TFRM) // 68352 values per batch
#define PI2      6.28318530717958647692f

typedef _Float16 v16h __attribute__((ext_vector_type(16)));
typedef float    v8f  __attribute__((ext_vector_type(8)));

// ---------------------------------------------------------------------------
// Twiddle B-fragment element mapping for v_wmma_f32_16x16x32_f16 (wave32).
// Fragment layout mirrors the documented 16-bit A layout (ISA 7.12.2):
//   lanes 0-15 : N = lane,     VGPR v<4 packs K = 2v,2v+1 ; v>=4 packs K = 16+2(v-4)+{0,1}
//   lanes16-31 : N = lane-16,  same but K offset +8
// ---------------------------------------------------------------------------
__device__ __forceinline__ int frag_k(int v, int pos, int hf) {
    return ((v < 4) ? (2 * v) : (16 + 2 * (v - 4))) + pos + 8 * hf;
}

// ---------------------------------------------------------------------------
// Kernel 0: build twiddle tables (pre-swizzled hi/lo f16 fragments),
// window table, and zero the per-batch valid counters.
// gid = ((nt*32 + kc)*32 + lane)*16 + j  for nt in [0,32), kc in [0,32)
// nt < 16  -> columns are Re bins nt*16+col  : cos(2*pi*bin*k/1024)
// nt >= 16 -> columns are Im bins (nt-16)*16+col : -sin(2*pi*bin*k/1024)
// ---------------------------------------------------------------------------
__global__ __launch_bounds__(256) void init_tables(_Float16* __restrict__ Bhi,
                                                   _Float16* __restrict__ Blo,
                                                   float* __restrict__ win,
                                                   int* __restrict__ nvalid) {
    int gid  = blockIdx.x * 256 + threadIdx.x;      // 0 .. 524287
    int j    = gid & 15;
    int lane = (gid >> 4) & 31;
    int kc   = (gid >> 9) & 31;
    int nt   = gid >> 14;

    int col = lane & 15;
    int hf  = lane >> 4;
    int v   = j >> 1;
    int pos = j & 1;
    int k   = kc * 32 + frag_k(v, pos, hf);
    int bin = (nt & 15) * 16 + col;
    int ph  = (bin * k) & (KFFT - 1);               // exact integer phase reduction
    float th = (float)ph * (PI2 / (float)KFFT);
    float val = (nt < 16) ? cosf(th) : -sinf(th);
    _Float16 h = (_Float16)val;
    Bhi[gid] = h;
    Blo[gid] = (_Float16)(val - (float)h);

    if (gid < KFFT) {
        float w = 0.0f;
        int off = (KFFT - MWIN) / 2;                // 112
        if (gid >= off && gid < off + MWIN) {
            float n = (float)(gid - off);
            w = 0.5f - 0.5f * cosf(PI2 * n / (float)MWIN);   // periodic hann
        }
        win[gid] = w;
    }
    if (gid < BATCH) nvalid[gid] = 0;
}

// ---------------------------------------------------------------------------
// Split-f16 WMMA: acc += (a_hi+a_lo) * (b_hi+b_lo), dropping a_lo*b_lo.
// ---------------------------------------------------------------------------
__device__ __forceinline__ v8f tile_mma(v8f acc, v16h a_hi, v16h a_lo,
                                        const _Float16* __restrict__ Bhi,
                                        const _Float16* __restrict__ Blo,
                                        int nt, int kc, int lane) {
    const size_t off = (size_t)(((nt * 32) + kc) * 32 + lane) * 16;
    const v16h bh = *(const v16h*)(Bhi + off);
    const v16h bl = *(const v16h*)(Blo + off);
    acc = __builtin_amdgcn_wmma_f32_16x16x32_f16(false, a_hi, false, bh, (short)0, acc, false, false);
    acc = __builtin_amdgcn_wmma_f32_16x16x32_f16(false, a_lo, false, bh, (short)0, acc, false, false);
    acc = __builtin_amdgcn_wmma_f32_16x16x32_f16(false, a_hi, false, bl, (short)0, acc, false, false);
    return acc;
}

// ---------------------------------------------------------------------------
// Kernel 1: STFT power spectrogram via DFT matmul on WMMA.
// Grid: 534 M-tiles (16 frames each). Block: 256 thr = 8 waves.
// Wave w owns N-tiles {2w, 2w+1} (Re) and {2w+16, 2w+17} (Im) so that
// |X|^2 = Re^2 + Im^2 is lane-local on the C/D fragment layout.
// ---------------------------------------------------------------------------
__global__ __launch_bounds__(256) void stft_wmma(const float* __restrict__ y,
                                                 const _Float16* __restrict__ Bhi,
                                                 const _Float16* __restrict__ Blo,
                                                 const float* __restrict__ win,
                                                 float* __restrict__ P) {
    __shared__ unsigned int Ahi[16][16];   // [row][kpair] packed 2 x f16
    __shared__ unsigned int Alo[16][16];

    const int mtile = blockIdx.x;
    const int tid   = threadIdx.x;
    const int lane  = tid & 31;
    const int wv    = tid >> 5;            // wave id 0..7
    const int col   = lane & 15;
    const int hf    = lane >> 4;

    // staging assignment: one (row, k-pair) per thread
    const int r_st  = tid >> 4;            // 0..15
    const int kp_st = tid & 15;            // pair index 0..15
    const int m_st  = mtile * 16 + r_st;
    const int b_st  = m_st / TFRM;
    const int fr_st = m_st - b_st * TFRM;
    const size_t ybase = (size_t)b_st * TLEN;

    v8f accRe0 = {}; v8f accRe1 = {}; v8f accIm0 = {}; v8f accIm1 = {};

    for (int kc = 0; kc < 32; ++kc) {
        // ---- stage windowed, reflect-padded A chunk into LDS (hi/lo split) ----
        unsigned int packH, packL;
        {
            float xs[2];
#pragma unroll
            for (int q = 0; q < 2; ++q) {
                int k = kc * 32 + kp_st * 2 + q;
                int p = fr_st * HOPSZ + k - (KFFT / 2);       // reflect pad by 512
                p = (p < 0) ? -p : p;
                p = (p >= TLEN) ? (2 * TLEN - 2 - p) : p;
                xs[q] = y[ybase + p] * win[k];
            }
            union { _Float16 h[2]; unsigned int u; } uh, ul;
            uh.h[0] = (_Float16)xs[0];
            uh.h[1] = (_Float16)xs[1];
            ul.h[0] = (_Float16)(xs[0] - (float)uh.h[0]);
            ul.h[1] = (_Float16)(xs[1] - (float)uh.h[1]);
            packH = uh.u; packL = ul.u;
        }
        __syncthreads();                    // previous fragments consumed
        Ahi[r_st][kp_st] = packH;
        Alo[r_st][kp_st] = packL;
        __syncthreads();

        // ---- build A fragments per the documented 16-bit A layout ----
        v16h a_hi, a_lo;
        {
            union { v16h v; unsigned int u[8]; } uh, ul;
#pragma unroll
            for (int vv = 0; vv < 8; ++vv) {
                int kbase = frag_k(vv, 0, hf);
                uh.u[vv] = Ahi[col][kbase >> 1];
                ul.u[vv] = Alo[col][kbase >> 1];
            }
            a_hi = uh.v; a_lo = ul.v;
        }

        // ---- 12 WMMAs: 4 n-tiles x 3 split terms ----
        accRe0 = tile_mma(accRe0, a_hi, a_lo, Bhi, Blo, 2 * wv,      kc, lane);
        accRe1 = tile_mma(accRe1, a_hi, a_lo, Bhi, Blo, 2 * wv + 1,  kc, lane);
        accIm0 = tile_mma(accIm0, a_hi, a_lo, Bhi, Blo, 2 * wv + 16, kc, lane);
        accIm1 = tile_mma(accIm1, a_hi, a_lo, Bhi, Blo, 2 * wv + 17, kc, lane);
    }

    // ---- power = Re^2 + Im^2 (lane-local) and scatter to P[b][bin][t] ----
#pragma unroll
    for (int i = 0; i < 8; ++i) {
        int M  = i + 8 * hf;                // C/D layout: lanes>=16 hold rows 8..15
        int m  = mtile * 16 + M;
        int b  = m / TFRM;
        int fr = m - b * TFRM;
        float p0 = accRe0[i] * accRe0[i] + accIm0[i] * accIm0[i];
        float p1 = accRe1[i] * accRe1[i] + accIm1[i] * accIm1[i];
        int bin0 = 32 * wv + col;
        P[(size_t)(b * NBINS + bin0)      * TFRM + fr] = p0;
        P[(size_t)(b * NBINS + bin0 + 16) * TFRM + fr] = p1;
    }
}

// ---------------------------------------------------------------------------
// Kernel 2: per-(batch,bin) decay detection + regression. One wave per row.
// Writes RT60 (or 1e30) in place over the spectrogram buffer, counts valid.
// ---------------------------------------------------------------------------
__global__ __launch_bounds__(256) void rt60_rows(const float* __restrict__ Pg,
                                                 float* __restrict__ vals,
                                                 int* __restrict__ nvalid) {
    __shared__ float Pr[8][280];
    const int tid  = threadIdx.x;
    const int lane = tid & 31;
    const int wv   = tid >> 5;
    const int row  = blockIdx.x * 8 + wv;          // 0..8191  (= b*256 + f)
    const float* src = Pg + (size_t)row * TFRM;

    for (int i = lane; i < TFRM; i += 32) Pr[wv][i] = src[i];
    if (lane < 13) Pr[wv][TFRM + lane] = src[TFRM - 1];   // clip-pad
    __syncthreads();

    // decreasing-run length starting at each t, and best window length L
    int Rarr[9];
    int cand = 0;
#pragma unroll
    for (int i = 0; i < 9; ++i) {
        int t = lane + 32 * i;
        int R = 0;
        if (t < TFRM) {
            for (int q = 0; q < 12; ++q) {
                if (Pr[wv][t + q + 1] < Pr[wv][t + q]) R++; else break;
            }
            int maxL = min(min(R + 1, TFRM - t), L_MAXC);
            if (maxL >= 3 && maxL > cand) cand = maxL;
        }
        Rarr[i] = R;
    }
#pragma unroll
    for (int s = 16; s > 0; s >>= 1) cand = max(cand, __shfl_xor(cand, s, 32));
    const int  lenL    = cand;
    const bool has_any = (lenL >= 3);

    const float fn  = (float)lenL;
    const float xm  = 0.5f * (fn - 1.0f);
    const float den = fn * (fn * fn - 1.0f) * (1.0f / 12.0f);

    int cnt = 0;
#pragma unroll
    for (int i = 0; i < 9; ++i) {
        int t = lane + 32 * i;
        if (t >= TFRM) continue;
        float val = 1e30f;
        if (has_any && (Rarr[i] >= lenL - 1) && (t + lenL <= TFRM)) {
            // EDC (reverse cumsum of masked power)
            float e[L_MAXC];
            float su = 0.0f;
#pragma unroll
            for (int q = L_MAXC - 1; q >= 0; --q) {
                su += (q < lenL) ? Pr[wv][t + q] : 0.0f;
                e[q] = su;
            }
            float db0 = 10.0f * log10f(fmaxf(e[0], 1e-10f));
            float sy = 0.0f, sxy = 0.0f, lastdb = 0.0f;
#pragma unroll
            for (int q = 0; q < L_MAXC; ++q) {
                if (q < lenL) {
                    float db = 10.0f * log10f(fmaxf(e[q], 1e-10f)) - db0;
                    sy  += db;
                    sxy += (float)q * db;
                    if (q == lenL - 1) lastdb = db;
                }
            }
            if (lastdb < -10.0f) {
                float slope = (sxy - xm * sy) / den;
                val = (-60.0f / slope) * ((float)HOPSZ / (float)FS_);
                cnt++;
            }
        }
        vals[(size_t)row * TFRM + t] = val;
    }
#pragma unroll
    for (int s = 16; s > 0; s >>= 1) cnt += __shfl_xor(cnt, s, 32);
    if (lane == 0 && cnt) atomicAdd(&nvalid[row >> 8], cnt);
}

// ---------------------------------------------------------------------------
// Kernel 3: exact lower-median per batch via 31-step binary search on float
// bit patterns (all candidates positive => uint order == float order),
// then affine correction + clamp.
// ---------------------------------------------------------------------------
__global__ __launch_bounds__(256) void median_out(const float* __restrict__ vals,
                                                  const int* __restrict__ nvalid,
                                                  const float* __restrict__ coeffs,
                                                  float* __restrict__ out) {
    __shared__ int red[256];
    const int b   = blockIdx.x;
    const int tid = threadIdx.x;
    const int nv  = nvalid[b];
    const float* vb = vals + (size_t)b * ROWCNT;
    const int kth = (nv > 0) ? ((nv - 1) >> 1) : 0;

    unsigned int lo = 0u, hi = 0x7F800000u;       // [0, +inf]
    for (int it = 0; it < 31; ++it) {
        unsigned int mid = (lo + hi) >> 1;
        float fm = __uint_as_float(mid);
        int c = 0;
        for (int idx = tid; idx < ROWCNT; idx += 256) c += (vb[idx] <= fm) ? 1 : 0;
        red[tid] = c;
        __syncthreads();
#pragma unroll
        for (int s = 128; s > 0; s >>= 1) {
            if (tid < s) red[tid] += red[tid + s];
            __syncthreads();
        }
        int total = red[0];
        __syncthreads();
        if (total >= kth + 1) hi = mid; else lo = mid + 1;
    }
    if (tid == 0) {
        float med = __uint_as_float(hi);
        float o = (nv > 0) ? (coeffs[0] + coeffs[1] * med) : 0.5f;
        out[b] = fmaxf(o, 0.01f);
    }
}

// ---------------------------------------------------------------------------
// Launcher. Workspace layout (all offsets 4 KiB aligned):
//   [0, 1MB)        Bhi twiddle fragments (524288 f16)
//   [1MB, 2MB)      Blo twiddle fragments
//   [2MB, +4KB)     window (1024 f32)
//   [2MB+4KB, ...)  nvalid (32 i32)
//   [2MB+8KB, ...)  P / vals (32*256*267 f32 = 8.34 MB, reused in place)
// ---------------------------------------------------------------------------
extern "C" void kernel_launch(void* const* d_in, const int* in_sizes, int n_in,
                              void* d_out, int out_size, void* d_ws, size_t ws_size,
                              hipStream_t stream) {
    const float* y      = (const float*)d_in[0];
    const float* coeffs = (const float*)d_in[1];
    float* out          = (float*)d_out;

    char* ws = (char*)d_ws;
    _Float16* Bhi = (_Float16*)(ws);
    _Float16* Blo = (_Float16*)(ws + (1u << 20));
    float*    win = (float*)(ws + (2u << 20));
    int*   nvalid = (int*)(ws + (2u << 20) + 4096);
    float*      P = (float*)(ws + (2u << 20) + 8192);

    init_tables<<<2048, 256, 0, stream>>>(Bhi, Blo, win, nvalid);
    stft_wmma<<<MTILES, 256, 0, stream>>>(y, Bhi, Blo, win, P);
    rt60_rows<<<(BATCH * NBINS) / 8, 256, 0, stream>>>(P, P, nvalid);
    median_out<<<BATCH, 256, 0, stream>>>(P, nvalid, coeffs, out);
}
